// GCNNetGraphCls_695784702038
// MI455X (gfx1250) — matrix operand order repaired and checked by reading, not compiled, not run on hardware
//
#include <hip/hip_runtime.h>
#include <math.h>

typedef float v2f  __attribute__((ext_vector_type(2)));
typedef float v8f  __attribute__((ext_vector_type(8)));
typedef _Float16 v16h __attribute__((ext_vector_type(16)));

#ifndef __has_builtin
#define __has_builtin(x) 0
#endif

#if __has_builtin(__builtin_amdgcn_wmma_f32_16x16x4_f32)
#define HAVE_WMMA_F32_X4 1
#else
#define HAVE_WMMA_F32_X4 0
#endif

// ---------------------------------------------------------------------------
// Utility: zero a float region
// ---------------------------------------------------------------------------
__global__ void gcn_zero_kernel(float* __restrict__ p, long n) {
    long i = (long)blockIdx.x * blockDim.x + threadIdx.x;
    if (i < n) p[i] = 0.0f;
}

// ---------------------------------------------------------------------------
// deg[row[e]] += 1 over edges (self-loop folded in later as +1)
// ---------------------------------------------------------------------------
__global__ void gcn_deg_kernel(const int* __restrict__ rows, float* __restrict__ deg, int E) {
    int e = blockIdx.x * blockDim.x + threadIdx.x;
    if (e < E) atomicAdd(&deg[rows[e]], 1.0f);
}

// counts[batch[i]] += 1 over nodes
__global__ void gcn_counts_kernel(const int* __restrict__ batch, float* __restrict__ counts, int N) {
    int i = blockIdx.x * blockDim.x + threadIdx.x;
    if (i < N) atomicAdd(&counts[batch[i]], 1.0f);
}

// dinv[i] = rsqrt(deg[i] + 1)   (+1 = self loop; deg>=1 always)
__global__ void gcn_dinv_kernel(const float* __restrict__ deg, float* __restrict__ dinv, int N) {
    int i = blockIdx.x * blockDim.x + threadIdx.x;
    if (i < N) dinv[i] = rsqrtf(deg[i] + 1.0f);
}

// ---------------------------------------------------------------------------
// WMMA GEMM: out[N,64] = H[N,64] @ W[64,64]
// One wave per 16-row tile; 4 column tiles of 16; K swept by WMMA.
// Wave-uniform guard keeps EXEC all-ones inside active waves (WMMA req).
// ---------------------------------------------------------------------------
__global__ __launch_bounds__(256) void gcn_gemm64_kernel(
    const float* __restrict__ H, const float* __restrict__ W,
    float* __restrict__ out, int nrows) {
    const int lane = threadIdx.x & 31;
    const int wave = threadIdx.x >> 5;
    const int tile = blockIdx.x * 8 + wave;
    if (tile * 16 >= nrows) return;          // uniform per wave

    const int row0 = tile * 16;
    const int r = lane & 15;
    const float* __restrict__ hrow = H + (long)(row0 + r) * 64;

    v8f acc[4] = {};

#if HAVE_WMMA_F32_X4
    // fp32 WMMA 16x16x4: A lanes0-15 hold K{0,1}, lanes16-31 hold K{2,3}
    const int kb = (lane < 16) ? 0 : 2;
    for (int kk = 0; kk < 16; ++kk) {
        const int k0 = kk * 4;
        v2f a;
        a.x = hrow[k0 + kb + 0];
        a.y = hrow[k0 + kb + 1];
#pragma unroll
        for (int t = 0; t < 4; ++t) {
            const int n = t * 16 + r;
            v2f b;
            b.x = W[(k0 + kb + 0) * 64 + n];
            b.y = W[(k0 + kb + 1) * 64 + n];
            acc[t] = __builtin_amdgcn_wmma_f32_16x16x4_f32(
                false, a, false, b, (short)0, acc[t], false, false);
        }
    }
#else
    // f16-in / f32-accumulate fallback (codegen-confirmed builtin)
    const int kbA = (lane < 16) ? 0 : 8;    // A: elems 0-7 -> K kbA+e, 8-15 -> K kbA+16+(e-8)
    const int kbB = (lane < 16) ? 0 : 8;    // B: elems 0-7 -> K kbB+e, 8-15 -> K kbB+16+(e-8)
    for (int kk = 0; kk < 2; ++kk) {
        const int k0 = kk * 32;
        v16h a;
#pragma unroll
        for (int e = 0; e < 8; ++e) {
            a[e]     = (_Float16)hrow[k0 + kbA + e];
            a[8 + e] = (_Float16)hrow[k0 + kbA + 16 + e];
        }
#pragma unroll
        for (int t = 0; t < 4; ++t) {
            const int n = t * 16 + r;
            v16h b;
#pragma unroll
            for (int e = 0; e < 16; ++e)
                b[e] = (_Float16)W[(k0 + kbB + e + (e >= 8 ? 8 : 0)) * 64 + n];
            acc[t] = __builtin_amdgcn_wmma_f32_16x16x32_f16(
                false, a, false, b, (short)0, acc[t], false, false);
        }
    }
#endif

    // C/D layout: lane holds column n = lane&15; VGPR v holds row m = v + (lane<16?0:8)
    const int mb = (lane < 16) ? 0 : 8;
#pragma unroll
    for (int t = 0; t < 4; ++t) {
        const int n = t * 16 + r;
#pragma unroll
        for (int v = 0; v < 8; ++v)
            out[(long)(row0 + mb + v) * 64 + n] = acc[t][v];
    }
}

// ---------------------------------------------------------------------------
// Edge aggregation: agg[row] += dinv[row]*dinv[col] * hW[col]
// 16 lanes per edge, 4 channels per lane (float4 gather, f32 atomic scatter).
// ---------------------------------------------------------------------------
__global__ __launch_bounds__(256) void gcn_edge_agg_kernel(
    const float* __restrict__ hW, const int* __restrict__ rows,
    const int* __restrict__ cols, const float* __restrict__ dinv,
    float* __restrict__ agg, int E) {
    long tid = (long)blockIdx.x * blockDim.x + threadIdx.x;
    long e = tid >> 4;
    if (e >= E) return;
    int q = (int)(tid & 15) * 4;
    int rr = rows[e];
    int cc = cols[e];
    float nm = dinv[rr] * dinv[cc];
    const float4 v = *(const float4*)(hW + (long)cc * 64 + q);
    float* dst = agg + (long)rr * 64 + q;
    atomicAdd(dst + 0, nm * v.x);
    atomicAdd(dst + 1, nm * v.y);
    atomicAdd(dst + 2, nm * v.z);
    atomicAdd(dst + 3, nm * v.w);
}

// ---------------------------------------------------------------------------
// Fused: h = relu(agg + dinv^2 * hW (self loop) + bias); pool[batch] += h
// In-place on agg.
// ---------------------------------------------------------------------------
__global__ __launch_bounds__(256) void gcn_bias_relu_pool_kernel(
    float* __restrict__ agg, const float* __restrict__ hW,
    const float* __restrict__ dinv, const float* __restrict__ bias,
    const int* __restrict__ batch, float* __restrict__ pool, int N) {
    long idx = (long)blockIdx.x * blockDim.x + threadIdx.x;
    if (idx >= (long)N * 64) return;
    int node = (int)(idx >> 6);
    int ch = (int)(idx & 63);
    float di = dinv[node];
    float v = agg[idx] + di * di * hW[idx] + bias[ch];
    v = fmaxf(v, 0.0f);
    agg[idx] = v;
    atomicAdd(&pool[(long)batch[node] * 64 + ch], v);
}

// ---------------------------------------------------------------------------
// Head: out = log_softmax(relu((pool/denom) @ Wl + bl)), one block per graph.
// pool already holds segsum(h1)+segsum(h2) so pool/denom = x1+x2.
// ---------------------------------------------------------------------------
__global__ __launch_bounds__(64) void gcn_head_kernel(
    const float* __restrict__ pool, const float* __restrict__ counts,
    const float* __restrict__ Wl, const float* __restrict__ bl,
    float* __restrict__ out, int n_cls) {
    __shared__ float s[64];
    __shared__ float lg[32];
    int g = blockIdx.x;
    int t = threadIdx.x;
    float denom = fmaxf(counts[g], 1.0f);
    s[t] = pool[(long)g * 64 + t] / denom;
    __syncthreads();
    if (t < n_cls) {
        float a = bl[t];
        for (int k = 0; k < 64; ++k) a += s[k] * Wl[k * n_cls + t];
        lg[t] = fmaxf(a, 0.0f);
    }
    __syncthreads();
    if (t < n_cls) {
        float mx = lg[0];
        for (int i = 1; i < n_cls; ++i) mx = fmaxf(mx, lg[i]);
        float sum = 0.0f;
        for (int i = 0; i < n_cls; ++i) sum += expf(lg[i] - mx);
        out[(long)g * n_cls + t] = lg[t] - mx - logf(sum);
    }
}

// ---------------------------------------------------------------------------
// Launch orchestration
// ---------------------------------------------------------------------------
static inline unsigned cdivu(long a, long b) { return (unsigned)((a + b - 1) / b); }

extern "C" void kernel_launch(void* const* d_in, const int* in_sizes, int n_in,
                              void* d_out, int out_size, void* d_ws, size_t ws_size,
                              hipStream_t stream) {
    (void)n_in; (void)ws_size;
    const float* x    = (const float*)d_in[0];
    const int*   ei   = (const int*)d_in[1];
    const int*   bat  = (const int*)d_in[2];
    const float* W1   = (const float*)d_in[3];
    const float* b1   = (const float*)d_in[4];
    const float* W2   = (const float*)d_in[5];
    const float* b2   = (const float*)d_in[6];
    const float* Wl   = (const float*)d_in[7];
    const float* bl   = (const float*)d_in[8];
    float* out = (float*)d_out;

    const int D = 64;
    const int N = in_sizes[0] / D;
    const int E = in_sizes[1] / 2;
    const int n_cls = 10;
    const int G = out_size / n_cls;

    const int* rows = ei;       // edge_index[0] = targets (scatter dst)
    const int* cols = ei + E;   // edge_index[1] = sources (gather src)

    // Workspace layout (floats); zero-init regions contiguous up front.
    float* ws     = (float*)d_ws;
    float* deg    = ws;                               // N
    float* counts = deg + N;                          // G
    float* pool   = counts + G;                       // G*64
    float* agg    = pool + (long)G * D;               // N*64  (h / h2 buffer)
    float* dinv   = agg + (long)N * D;                // N
    float* bufA   = dinv + N;                         // N*64  (hW buffer)

    const int B = 256;

    // 1) zero deg|counts|pool|agg in one shot
    long z1 = (long)N + G + (long)G * D + (long)N * D;
    gcn_zero_kernel<<<cdivu(z1, B), B, 0, stream>>>(deg, z1);

    // 2) degrees, graph counts, dinv
    gcn_deg_kernel<<<cdivu(E, B), B, 0, stream>>>(rows, deg, E);
    gcn_counts_kernel<<<cdivu(N, B), B, 0, stream>>>(bat, counts, N);
    gcn_dinv_kernel<<<cdivu(N, B), B, 0, stream>>>(deg, dinv, N);

    const unsigned gemm_blocks = cdivu((N + 15) / 16, 8);
    const unsigned edge_blocks = cdivu((long)E * 16, B);
    const unsigned elem_blocks = cdivu((long)N * D, B);

    // 3) conv1: hW = x @ W1 ; agg over edges ; fused selfloop+bias+relu+pool
    gcn_gemm64_kernel<<<gemm_blocks, B, 0, stream>>>(x, W1, bufA, N);
    gcn_edge_agg_kernel<<<edge_blocks, B, 0, stream>>>(bufA, rows, cols, dinv, agg, E);
    gcn_bias_relu_pool_kernel<<<elem_blocks, B, 0, stream>>>(agg, bufA, dinv, b1, bat, pool, N);

    // 4) conv2: hW = h @ W2 ; re-zero agg ; aggregate ; fused pass
    gcn_gemm64_kernel<<<gemm_blocks, B, 0, stream>>>(agg, W2, bufA, N);
    gcn_zero_kernel<<<cdivu((long)N * D, B), B, 0, stream>>>(agg, (long)N * D);
    gcn_edge_agg_kernel<<<edge_blocks, B, 0, stream>>>(bufA, rows, cols, dinv, agg, E);
    gcn_bias_relu_pool_kernel<<<elem_blocks, B, 0, stream>>>(agg, bufA, dinv, b2, bat, pool, N);

    // 5) classifier head + log_softmax
    gcn_head_kernel<<<G, 64, 0, stream>>>(pool, counts, Wl, bl, out, n_cls);
}
